// MSA_20383914787427
// MI455X (gfx1250) — matrix operand order repaired and checked by reading, not compiled, not run on hardware
//
#include <hip/hip_runtime.h>

// ---- problem dims ----
#define B_     4
#define SEQ    2048
#define DIM    1024
#define NHEADS 16
#define DHEAD  64
#define BN     (B_ * SEQ)   // 8192 rows
#define E3     (3 * DIM)    // 3072 qkv outputs

typedef __attribute__((ext_vector_type(8)))  float  v8f;
typedef __attribute__((ext_vector_type(16))) __bf16 bf16x16;
typedef __attribute__((ext_vector_type(8)))  __bf16 bf16x8;
typedef __attribute__((ext_vector_type(4)))  unsigned int u32x4;
typedef __attribute__((ext_vector_type(8)))  int i32x8;
typedef __attribute__((ext_vector_type(4)))  int i32x4;

// ---------------- fragment helpers (16-bit WMMA operands, this code's contiguous k = 16*half + s order) -------------
__device__ __forceinline__ bf16x8 ld8(const __bf16* p) { return *(const bf16x8*)p; }

__device__ __forceinline__ bf16x16 cat16(bf16x8 lo, bf16x8 hi) {
  bf16x16 r;
#pragma unroll
  for (int i = 0; i < 8; ++i) { r[i] = lo[i]; r[i + 8] = hi[i]; }
  return r;
}

// A-matrix 16x32 bf16: lanes0-15 row=lane, K=0..15 contiguous; lanes16-31 K=16..31
__device__ __forceinline__ bf16x16 load_a(const __bf16* rowp, int half) {
  return cat16(ld8(rowp + half * 16), ld8(rowp + half * 16 + 8));
}

// B-matrix 32x16 bf16: lanes0-15 col=lane, K=0..15 contiguous; lanes16-31 K=16..31
__device__ __forceinline__ bf16x16 load_b(const __bf16* colp, int half) {
  return cat16(ld8(colp + half * 16), ld8(colp + half * 16 + 8));
}

__device__ __forceinline__ v8f wmma_bf16(bf16x16 a, bf16x16 b, v8f c) {
  return __builtin_amdgcn_wmma_f32_16x16x32_bf16(false, a, false, b, (short)0, c,
                                                 false, false);
}

// ---------------- Tensor Data Mover: 2-D bf16 tile -> LDS ----------------
// Builds D# group0/group1 per CDNA5 ISA (count=1, type=2 "image", data_size=2B)
// and issues TENSOR_LOAD_TO_LDS. Caller syncs with s_wait_tensorcnt + barrier.
__device__ __forceinline__ void tdm_load_2d(void* lds_ptr, const void* gptr,
                                            unsigned tile_x, unsigned tile_y,
                                            unsigned tensor_x, unsigned tensor_y,
                                            unsigned stride_x /*elements*/) {
  const unsigned lds_off = (unsigned)(unsigned long long)(uintptr_t)lds_ptr;
  const unsigned long long ga = (unsigned long long)(uintptr_t)gptr;
  u32x4 g0;
  g0[0] = 1u;                                     // count=1 (valid descriptor)
  g0[1] = lds_off;                                // lds_addr (bytes)
  g0[2] = (unsigned)(ga & 0xffffffffu);           // global_addr[31:0]
  g0[3] = (unsigned)((ga >> 32) & 0x1ffffffu)     // global_addr[56:32]
          | (2u << 30);                           // type=2
  i32x8 g1;
  g1[0] = (int)(1u << 16);                        // data_size=1 -> 2 bytes
  g1[1] = (int)((tensor_x & 0xffffu) << 16);      // tensor_dim0[15:0]
  g1[2] = (int)(((tensor_x >> 16) & 0xffffu) | ((tensor_y & 0xffffu) << 16));
  g1[3] = (int)(((tensor_y >> 16) & 0xffffu) | ((tile_x & 0xffffu) << 16));
  g1[4] = (int)(tile_y & 0xffffu);                // tile_dim1 (tile_dim2=0)
  g1[5] = (int)stride_x;                          // tensor_dim0_stride[31:0]
  g1[6] = 0;
  g1[7] = 0;
  const i32x4 z = {0, 0, 0, 0};
#if defined(__clang_major__) && (__clang_major__ >= 23)
  const i32x8 z8 = {0, 0, 0, 0, 0, 0, 0, 0};
  __builtin_amdgcn_tensor_load_to_lds(g0, g1, z, z, z8, 0);
#else
  __builtin_amdgcn_tensor_load_to_lds(g0, g1, z, z, 0);
#endif
}

// ---------------- kernel 0: fp32 -> bf16 conversion ----------------
__global__ void cvt_bf16(const float* __restrict__ s, __bf16* __restrict__ d, int n) {
  int i = blockIdx.x * blockDim.x + threadIdx.x;
  int stride = gridDim.x * blockDim.x;
  for (; i < n; i += stride) d[i] = (__bf16)s[i];
}

// ---------------- kernel 1: QKV GEMM  qkv = x * Wqkv^T + b -----------------
// Workgroup computes a 128x128 tile: TDM double-buffers A(128x32) and B(128x32)
// panels in LDS; each of 8 waves owns a 16-row strip x 128 cols (8 accumulators,
// 8 WMMAs per barrier, A fragment reused 8x).
__global__ __launch_bounds__(256) void qkv_gemm(
    const __bf16* __restrict__ xb, const __bf16* __restrict__ wb,
    const float* __restrict__ bias, __bf16* __restrict__ q,
    __bf16* __restrict__ k, __bf16* __restrict__ vT) {
  __shared__ __align__(16) __bf16 As[2][128 * 32];
  __shared__ __align__(16) __bf16 Bs[2][128 * 32];

  const int lane = threadIdx.x & 31;
  const int wave = threadIdx.x >> 5;
  const int l15 = lane & 15, half = lane >> 4;
  const int ET = E3 / 128;                  // 24 col blocks
  const int mt = blockIdx.x / ET, et = blockIdx.x % ET;
  const int row0 = mt * 128, e0 = et * 128;

  if (wave == 0) {
    tdm_load_2d(&As[0][0], xb + (size_t)row0 * DIM, 32, 128, DIM, BN, DIM);
    tdm_load_2d(&Bs[0][0], wb + (size_t)e0 * DIM, 32, 128, DIM, E3, DIM);
  }

  v8f acc[8] = {};
  for (int c = 0; c < DIM / 32; ++c) {
    const int cur = c & 1;
    if (wave == 0) __builtin_amdgcn_s_wait_tensorcnt(0);
    __syncthreads();
    if (wave == 0 && c + 1 < DIM / 32) {
      const int k0 = (c + 1) * 32;
      tdm_load_2d(&As[1 - cur][0], xb + (size_t)row0 * DIM + k0, 32, 128, DIM, BN, DIM);
      tdm_load_2d(&Bs[1 - cur][0], wb + (size_t)e0 * DIM + k0, 32, 128, DIM, E3, DIM);
    }
    const bf16x16 a = load_a(&As[cur][(wave * 16 + l15) * 32], half);
    bf16x16 bf[8];
#pragma unroll
    for (int nt = 0; nt < 8; ++nt)
      bf[nt] = load_b(&Bs[cur][(nt * 16 + l15) * 32], half);
#pragma unroll
    for (int nt = 0; nt < 8; ++nt) acc[nt] = wmma_bf16(a, bf[nt], acc[nt]);
  }

  const int which = e0 / DIM;               // 0:q 1:k 2:v (uniform: 1024 % 128 == 0)
#pragma unroll
  for (int nt = 0; nt < 8; ++nt) {
    const int e = e0 + nt * 16 + l15;
    const float bv = bias[e];
    const int eh = e - which * DIM;
    const int h = eh >> 6, dh = eh & 63;
#pragma unroll
    for (int r = 0; r < 8; ++r) {
      const int row = row0 + wave * 16 + r + 8 * half;   // C layout: M = r + 8*half
      const int bi = row >> 11, ni = row & (SEQ - 1);
      const float val = acc[nt][r] + bv;
      const size_t bh = (size_t)bi * NHEADS + h;
      if (which == 0)      q [(bh * SEQ + ni) * DHEAD + dh] = (__bf16)val;
      else if (which == 1) k [(bh * SEQ + ni) * DHEAD + dh] = (__bf16)val;
      else                 vT[(bh * DHEAD + dh) * SEQ + ni] = (__bf16)val;
    }
  }
}

// ---------------- kernel 2: flash attention ----------------
// All 8 waves of a block share one (b,h) head; 64-key K (64x64) and V^T (64x64)
// tiles are TDM double-buffered in LDS and shared by all waves (16 WMMA/barrier).
// S^T = K*Q^T puts softmax rows on lanes; O^T = V^T * P^T needs no P transpose.
__global__ __launch_bounds__(256) void flash_attn(
    const __bf16* __restrict__ q, const __bf16* __restrict__ k,
    const __bf16* __restrict__ vT, __bf16* __restrict__ y) {
  __shared__ __align__(16) __bf16 Kb[2][64 * 64];
  __shared__ __align__(16) __bf16 Vb[2][64 * 64];

  const int lane = threadIdx.x & 31;
  const int wave = threadIdx.x >> 5;
  const int l15 = lane & 15, half = lane >> 4;
  const int bh = blockIdx.x >> 4;              // 0..63 (= b*NHEADS + h)
  const int qt = (blockIdx.x & 15) * 8 + wave; // q tile 0..127 within head

  const __bf16* qp = q  + (size_t)bh * SEQ * DHEAD;
  const __bf16* kp = k  + (size_t)bh * SEQ * DHEAD;
  const __bf16* vp = vT + (size_t)bh * DHEAD * SEQ;

  // Q^T B-fragments (d-chunks 0..31 / 32..63); lane = q-row within tile
  const __bf16* qrow = qp + (size_t)(qt * 16 + l15) * DHEAD;
  const bf16x16 bq0 = load_b(qrow + 0, half);
  const bf16x16 bq1 = load_b(qrow + 32, half);

  if (wave == 0) {
    tdm_load_2d(&Kb[0][0], kp, 64, 64, DHEAD, SEQ, DHEAD);   // 64 key rows x 64 d
    tdm_load_2d(&Vb[0][0], vp, 64, 64, SEQ, DHEAD, SEQ);     // 64 d rows x 64 n
  }

  float mrun = -3.0e38f, lrun = 0.f;
  v8f o[4] = {};

  for (int nb = 0; nb < SEQ; nb += 64) {
    const int cur = (nb >> 6) & 1;
    if (wave == 0) __builtin_amdgcn_s_wait_tensorcnt(0);
    __syncthreads();
    if (wave == 0 && nb + 64 < SEQ) {
      tdm_load_2d(&Kb[1 - cur][0], kp + (size_t)(nb + 64) * DHEAD, 64, 64, DHEAD,
                  SEQ, DHEAD);
      tdm_load_2d(&Vb[1 - cur][0], vp + (nb + 64), 64, 64, SEQ, DHEAD, SEQ);
    }

    // ---- S^T: 4 key sub-tiles x (K=64 over two 32-chunks) ----
    const __bf16* Kc = &Kb[cur][0];
    v8f st[4] = {};
#pragma unroll
    for (int t = 0; t < 4; ++t) {
      const __bf16* krow = Kc + (t * 16 + l15) * 64;
      st[t] = wmma_bf16(load_a(krow + 0, half), bq0, st[t]);
      st[t] = wmma_bf16(load_a(krow + 32, half), bq1, st[t]);
    }

    // ---- online softmax (row = q index = lane&15, replicated across halves) ----
    const float sc = 0.125f;                 // Dh^-0.5
    float p[4][8];
    float mx = -3.0e38f;
#pragma unroll
    for (int t = 0; t < 4; ++t)
#pragma unroll
      for (int r = 0; r < 8; ++r) {
        p[t][r] = st[t][r] * sc;
        mx = fmaxf(mx, p[t][r]);
      }
    mx = fmaxf(mx, __shfl_xor(mx, 16, 32));
    const float mnew = fmaxf(mrun, mx);
    const float corr = __expf(mrun - mnew);
    float rsum = 0.f;
#pragma unroll
    for (int t = 0; t < 4; ++t)
#pragma unroll
      for (int r = 0; r < 8; ++r) {
        p[t][r] = __expf(p[t][r] - mnew);
        rsum += p[t][r];
      }
    rsum += __shfl_xor(rsum, 16, 32);
    lrun = lrun * corr + rsum;
    mrun = mnew;
#pragma unroll
    for (int dt = 0; dt < 4; ++dt)
#pragma unroll
      for (int r = 0; r < 8; ++r) o[dt][r] *= corr;

    // ---- P^T as two B-fragments (keys 0..31 / 32..63) via half-wave exchange ----
    float ex[4][8];
#pragma unroll
    for (int t = 0; t < 4; ++t)
#pragma unroll
      for (int r = 0; r < 8; ++r) ex[t][r] = __shfl_xor(p[t][r], 16, 32);
    bf16x16 bp0, bp1;
#pragma unroll
    for (int j = 0; j < 8; ++j) {
      bp0[j]     = (__bf16)(half ? ex[1][j] : p[0][j]);
      bp0[j + 8] = (__bf16)(half ? p[1][j] : ex[0][j]);
      bp1[j]     = (__bf16)(half ? ex[3][j] : p[2][j]);
      bp1[j + 8] = (__bf16)(half ? p[3][j] : ex[2][j]);
    }

    // ---- O^T += V^T * P^T : 4 d-tiles x 2 key chunks ----
    const __bf16* Vc = &Vb[cur][0];
#pragma unroll
    for (int dt = 0; dt < 4; ++dt) {
      const __bf16* vrow = Vc + (dt * 16 + l15) * 64;
      o[dt] = wmma_bf16(load_a(vrow + 0, half), bp0, o[dt]);
      o[dt] = wmma_bf16(load_a(vrow + 32, half), bp1, o[dt]);
    }
  }

  // ---- normalize + write y[b, n, h*64 + d] (contiguous 8-bf16 per tile) ----
  const float inv = 1.f / lrun;
  const int b = bh >> 4, h = bh & 15;
  __bf16* yrow = y + ((size_t)(b * SEQ + qt * 16 + l15)) * DIM + h * DHEAD;
#pragma unroll
  for (int dt = 0; dt < 4; ++dt) {
    bf16x8 pk;
#pragma unroll
    for (int r = 0; r < 8; ++r) pk[r] = (__bf16)(o[dt][r] * inv);
    *(bf16x8*)(yrow + dt * 16 + half * 8) = pk;
  }
}

// ---------------- kernel 3: output GEMM  out = y * Wout^T + b (fp32 out) ----------
__global__ __launch_bounds__(256) void out_gemm(
    const __bf16* __restrict__ yb, const __bf16* __restrict__ wb,
    const float* __restrict__ bias, float* __restrict__ out) {
  __shared__ __align__(16) __bf16 As[2][128 * 32];
  __shared__ __align__(16) __bf16 Bs[2][128 * 32];

  const int lane = threadIdx.x & 31;
  const int wave = threadIdx.x >> 5;
  const int l15 = lane & 15, half = lane >> 4;
  const int ET = DIM / 128;                 // 8 col blocks
  const int mt = blockIdx.x / ET, et = blockIdx.x % ET;
  const int row0 = mt * 128, e0 = et * 128;

  if (wave == 0) {
    tdm_load_2d(&As[0][0], yb + (size_t)row0 * DIM, 32, 128, DIM, BN, DIM);
    tdm_load_2d(&Bs[0][0], wb + (size_t)e0 * DIM, 32, 128, DIM, DIM, DIM);
  }

  v8f acc[8] = {};
  for (int c = 0; c < DIM / 32; ++c) {
    const int cur = c & 1;
    if (wave == 0) __builtin_amdgcn_s_wait_tensorcnt(0);
    __syncthreads();
    if (wave == 0 && c + 1 < DIM / 32) {
      const int k0 = (c + 1) * 32;
      tdm_load_2d(&As[1 - cur][0], yb + (size_t)row0 * DIM + k0, 32, 128, DIM, BN, DIM);
      tdm_load_2d(&Bs[1 - cur][0], wb + (size_t)e0 * DIM + k0, 32, 128, DIM, DIM, DIM);
    }
    const bf16x16 a = load_a(&As[cur][(wave * 16 + l15) * 32], half);
    bf16x16 bf[8];
#pragma unroll
    for (int nt = 0; nt < 8; ++nt)
      bf[nt] = load_b(&Bs[cur][(nt * 16 + l15) * 32], half);
#pragma unroll
    for (int nt = 0; nt < 8; ++nt) acc[nt] = wmma_bf16(a, bf[nt], acc[nt]);
  }

#pragma unroll
  for (int nt = 0; nt < 8; ++nt) {
    const int e = e0 + nt * 16 + l15;
    const float bv = bias[e];
#pragma unroll
    for (int r = 0; r < 8; ++r) {
      const int row = row0 + wave * 16 + r + 8 * half;
      out[(size_t)row * DIM + e] = acc[nt][r] + bv;
    }
  }
}

// ---------------- host launcher ----------------
extern "C" void kernel_launch(void* const* d_in, const int* in_sizes, int n_in,
                              void* d_out, int out_size, void* d_ws, size_t ws_size,
                              hipStream_t stream) {
  const float* x     = (const float*)d_in[0];
  const float* w_qkv = (const float*)d_in[1];
  const float* b_qkv = (const float*)d_in[2];
  const float* w_out = (const float*)d_in[3];
  const float* b_out = (const float*)d_in[4];
  float* out = (float*)d_out;

  char* ws = (char*)d_ws;
  size_t off = 0;
  __bf16* xb    = (__bf16*)(ws + off); off += (size_t)BN * DIM * 2;
  __bf16* wqkvb = (__bf16*)(ws + off); off += (size_t)E3 * DIM * 2;
  __bf16* woutb = (__bf16*)(ws + off); off += (size_t)DIM * DIM * 2;
  __bf16* qb    = (__bf16*)(ws + off); off += (size_t)BN * DIM * 2;
  __bf16* kb    = (__bf16*)(ws + off); off += (size_t)BN * DIM * 2;
  __bf16* vTb   = (__bf16*)(ws + off); off += (size_t)BN * DIM * 2;
  __bf16* yb    = (__bf16*)(ws + off); off += (size_t)BN * DIM * 2;

  cvt_bf16<<<4096, 256, 0, stream>>>(x, xb, BN * DIM);
  cvt_bf16<<<2048, 256, 0, stream>>>(w_qkv, wqkvb, E3 * DIM);
  cvt_bf16<<<1024, 256, 0, stream>>>(w_out, woutb, DIM * DIM);

  // 64 row-blocks * 24 col-blocks (128x128 per workgroup)
  qkv_gemm<<<1536, 256, 0, stream>>>(xb, wqkvb, b_qkv, qb, kb, vTb);
  // 64 heads * 16 blocks, 8 q-tiles per block
  flash_attn<<<1024, 256, 0, stream>>>(qb, kb, vTb, yb);
  // 64 row-blocks * 8 col-blocks
  out_gemm<<<512, 256, 0, stream>>>(yb, woutb, b_out, out);
}
